// CFMEncoder_80272938762375
// MI455X (gfx1250) — matrix operand order repaired and checked by reading, not compiled, not run on hardware
//
#include <hip/hip_runtime.h>
#include <math.h>

typedef __attribute__((ext_vector_type(16))) __bf16 v16bf;
typedef __attribute__((ext_vector_type(8)))  float  v8f;

#define DIMC   128
#define HEADC  8
#define EC     50000
#define NWC    5000
#define NTC    2000
#define NOC    10000
#define TILES  (EC / 16)      // 3125 (E divisible by 16)
#define KCN    12             // 384 / 32 K-chunks
#define NCGN   64             // 1024 / 16 N-chunks

static __device__ __forceinline__ __bf16 f2bf(float f) {
    unsigned u = __float_as_uint(f);
    u += 0x7FFFu + ((u >> 16) & 1u);            // RNE
    unsigned short s = (unsigned short)(u >> 16);
    return __builtin_bit_cast(__bf16, s);
}
static __device__ __forceinline__ float bf2f(__bf16 b) {
    unsigned short s = __builtin_bit_cast(unsigned short, b);
    return __uint_as_float(((unsigned)s) << 16);
}
// order-preserving float->uint mapping for atomicMax
static __device__ __forceinline__ unsigned sfmap(float f) {
    unsigned u = __float_as_uint(f);
    return (u & 0x80000000u) ? ~u : (u | 0x80000000u);
}
static __device__ __forceinline__ float sfunmap(unsigned u) {
    u = (u & 0x80000000u) ? (u & 0x7FFFFFFFu) : ~u;
    return __uint_as_float(u);
}

// ---------------- init: z_w = bcast(x_worker), z_t = bcast(x_task), z_o = option_x
__global__ void k_init(float* out, const float* xw, const float* xt, const float* xo) {
    size_t i = (size_t)blockIdx.x * 256 + threadIdx.x;
    const size_t tot = (size_t)(NWC + NTC + NOC) * DIMC;
    if (i >= tot) return;
    if (i < (size_t)NWC * DIMC)              out[i] = xw[i & 127];
    else if (i < (size_t)(NWC + NTC) * DIMC) out[i] = xt[i & 127];
    else                                     out[i] = xo[i - (size_t)(NWC + NTC) * DIMC];
}

__global__ void k_fill(unsigned* p, size_t n, unsigned v) {
    size_t i = (size_t)blockIdx.x * 256 + threadIdx.x;
    if (i < n) p[i] = v;
}

// ---------------- pass0: gather edge features, write A-fragment-swizzled bf16
// layout: zA[((tile*KCN + kc)*32 + lane)*16 + j],  row M = lane%16, K = kc*32 + (lane/16)*16 + j
__global__ void k_pass0(__bf16* zA, const float* zw, const float* zt, const float* zo,
                        const int* wid, const int* tidv, const int* oid) {
    int t = blockIdx.x * 256 + threadIdx.x;
    const int total = TILES * KCN * 32;
    if (t >= total) return;
    int lane = t & 31;
    int kc   = (t >> 5) % KCN;
    int tile = t / (KCN * 32);
    int e    = tile * 16 + (lane & 15);
    int lhi  = lane >> 4;
    const float* rows[3] = { zw + (size_t)wid[e] * DIMC,
                             zt + (size_t)tidv[e] * DIMC,
                             zo + (size_t)oid[e] * DIMC };
    v16bf frag;
#pragma unroll
    for (int j = 0; j < 16; ++j) {
        int K = kc * 32 + lhi * 16 + j;   // 0..383
        float v = rows[K >> 7][K & 127];
        frag[j] = f2bf(v);
    }
    ((v16bf*)zA)[(size_t)(tile * KCN + kc) * 32 + lane] = frag;
}

// ---------------- convert Wq/Wk [384,1024] fp32 -> bf16 B-fragment layout
// B layout: col N = lane%16, K = kc*32 + (lane/16)*16 + j
__global__ void k_convW(__bf16* Bq, __bf16* Bk, const float* Wq, const float* Wk) {
    int t = blockIdx.x * 256 + threadIdx.x;
    const int per = KCN * NCGN * 32;
    if (t >= 2 * per) return;
    const float* W = (t < per) ? Wq : Wk;
    __bf16*      B = (t < per) ? Bq : Bk;
    int u = t % per;
    int lane = u & 31;
    int ncg  = (u >> 5) % NCGN;
    int kc   = u / (32 * NCGN);
    int n  = ncg * 16 + (lane & 15);
    int kb = kc * 32 + (lane >> 4) * 16;
    v16bf frag;
#pragma unroll
    for (int j = 0; j < 16; ++j) frag[j] = f2bf(W[(size_t)(kb + j) * 1024 + n]);
    ((v16bf*)B)[(size_t)(kc * NCGN + ncg) * 32 + lane] = frag;
}

// ---------------- M_h = Wv[:, h*128:(h+1)*128] @ Wo[h*128:(h+1)*128, :], bf16 B-frag layout
__global__ void k_convM(__bf16* BM, const float* Wv, const float* Wo) {
    int t = blockIdx.x * 256 + threadIdx.x;
    const int total = HEADC * KCN * 8 * 32 * 16;  // 393216
    if (t >= total) return;
    int j    = t & 15;
    int lane = (t >> 4) & 31;
    int nc   = (t >> 9) & 7;
    int kc   = (t >> 12) % KCN;
    int h    = (t >> 12) / KCN;
    int krow = kc * 32 + (lane >> 4) * 16 + j;    // 0..383
    int ncol = nc * 16 + (lane & 15);             // 0..127
    float acc = 0.f;
    for (int d = 0; d < 128; ++d)
        acc += Wv[(size_t)krow * 1024 + h * 128 + d] * Wo[(size_t)(h * 128 + d) * 128 + ncol];
    BM[t] = f2bf(acc);
}

// ---------------- c2[h][n] = bv_h @ Wo_h
__global__ void k_c2(float* c2, const float* bv, const float* Wo) {
    int t = blockIdx.x * 256 + threadIdx.x;
    if (t >= HEADC * DIMC) return;
    int h = t >> 7, n = t & 127;
    float a = 0.f;
    for (int d = 0; d < 128; ++d) a += bv[h * 128 + d] * Wo[(size_t)(h * 128 + d) * 128 + n];
    c2[t] = a;
}

// ---------------- pass1: score[e,h] = (q.k)*scale via WMMA, bias in C accumulator
__global__ __launch_bounds__(256) void k_pass1(float* score, const __bf16* zA,
                                               const __bf16* Bq, const __bf16* Bk,
                                               const float* bq, const float* bk) {
    int wv = threadIdx.x >> 5, lane = threadIdx.x & 31;
    int tile = blockIdx.x * 8 + wv;
    if (tile >= TILES) return;                 // wave-uniform exit
    int lcol = lane & 15, lhi = lane >> 4;
    const v16bf* ap  = (const v16bf*)zA + (size_t)tile * KCN * 32;
    const v16bf* bqp = (const v16bf*)Bq;
    const v16bf* bkp = (const v16bf*)Bk;
    v16bf a[KCN];
#pragma unroll
    for (int kc = 0; kc < KCN; ++kc) a[kc] = ap[kc * 32 + lane];
    const float scale = 0.088388347648318447f;  // 1/sqrt(128)
    for (int h = 0; h < HEADC; ++h) {
        v8f prod = {0.f, 0.f, 0.f, 0.f, 0.f, 0.f, 0.f, 0.f};
        for (int nc = 0; nc < 8; ++nc) {
            int ncg = h * 8 + nc;
            float bql = bq[ncg * 16 + lcol];
            float bkl = bk[ncg * 16 + lcol];
            v8f qa, ka;
#pragma unroll
            for (int r = 0; r < 8; ++r) { qa[r] = bql; ka[r] = bkl; }
#pragma unroll
            for (int kc = 0; kc < KCN; ++kc) {
                v16bf bw = bqp[(size_t)(kc * NCGN + ncg) * 32 + lane];
                qa = __builtin_amdgcn_wmma_f32_16x16x32_bf16(false, a[kc], false, bw,
                                                             (short)0, qa, false, false);
            }
#pragma unroll
            for (int kc = 0; kc < KCN; ++kc) {
                v16bf bw = bkp[(size_t)(kc * NCGN + ncg) * 32 + lane];
                ka = __builtin_amdgcn_wmma_f32_16x16x32_bf16(false, a[kc], false, bw,
                                                             (short)0, ka, false, false);
            }
#pragma unroll
            for (int r = 0; r < 8; ++r) prod[r] += qa[r] * ka[r];
        }
        // sum over the 16 columns (lanes within each 16-lane half)
#pragma unroll
        for (int r = 0; r < 8; ++r) {
            float v = prod[r];
            v += __shfl_xor(v, 1, 32);
            v += __shfl_xor(v, 2, 32);
            v += __shfl_xor(v, 4, 32);
            v += __shfl_xor(v, 8, 32);
            prod[r] = v;
        }
        if (lcol == 0) {
#pragma unroll
            for (int r = 0; r < 8; ++r)
                score[(size_t)(tile * 16 + lhi * 8 + r) * HEADC + h] = prod[r] * scale;
        }
    }
}

// ---------------- segment softmax pieces
__global__ void k_segmax(unsigned* m, const float* score, const int* idx) {
    int t = blockIdx.x * 256 + threadIdx.x;
    if (t >= EC * HEADC) return;
    int e = t >> 3, h = t & 7;
    atomicMax(&m[(size_t)idx[e] * HEADC + h], sfmap(score[t]));
}
__global__ void k_expsum(float* score, float* s, const unsigned* m, const int* idx) {
    int t = blockIdx.x * 256 + threadIdx.x;
    if (t >= EC * HEADC) return;
    int e = t >> 3, h = t & 7;
    float mm = sfunmap(m[(size_t)idx[e] * HEADC + h]);
    float ex = __expf(score[t] - mm);
    score[t] = ex;
    unsafeAtomicAdd(&s[(size_t)idx[e] * HEADC + h], ex);
}
__global__ void k_attn(float* score, const float* s, const int* idx) {
    int t = blockIdx.x * 256 + threadIdx.x;
    if (t >= EC * HEADC) return;
    int e = t >> 3, h = t & 7;
    score[t] = score[t] / (s[(size_t)idx[e] * HEADC + h] + 1e-16f);
}

// ---------------- pass3: agg[idx[e]] += sum_h attn_h * (z @ M_h) + bias terms
__global__ __launch_bounds__(256) void k_pass3(float* agg, const float* attn,
                                               const __bf16* zA, const __bf16* BM,
                                               const float* c2, const float* bo,
                                               const int* idx) {
    __shared__ float attnS[8][16 * HEADC];
    __shared__ int   idxS[8][16];
    int wv = threadIdx.x >> 5, lane = threadIdx.x & 31;
    int tile = blockIdx.x * 8 + wv;
    bool active = tile < TILES;
    if (active && lane < 16) {
        int e = tile * 16 + lane;
        idxS[wv][lane] = idx[e];
        for (int h = 0; h < HEADC; ++h) attnS[wv][lane * HEADC + h] = attn[(size_t)e * HEADC + h];
    }
    __syncthreads();
    if (!active) return;
    int lcol = lane & 15, lhi = lane >> 4;
    const v16bf* ap  = (const v16bf*)zA + (size_t)tile * KCN * 32;
    const v16bf* bmp = (const v16bf*)BM;
    v8f acc[8];
#pragma unroll
    for (int nc = 0; nc < 8; ++nc) {
        int col = nc * 16 + lcol;
        float base = bo[col];
#pragma unroll
        for (int r = 0; r < 8; ++r) {
            int M = r + lhi * 8;
            float v = base;
#pragma unroll
            for (int h = 0; h < HEADC; ++h) v += attnS[wv][M * HEADC + h] * c2[h * 128 + col];
            acc[nc][r] = v;
        }
    }
    for (int h = 0; h < HEADC; ++h) {
        // per-lane row scale (A-frag row = lane%16), done as packed bf16 math (v_pk_mul_bf16)
        __bf16 awb = f2bf(attnS[wv][lcol * HEADC + h]);
        v16bf aws;
#pragma unroll
        for (int j = 0; j < 16; ++j) aws[j] = awb;
        v16bf sa[KCN];
#pragma unroll
        for (int kc = 0; kc < KCN; ++kc) sa[kc] = ap[kc * 32 + lane] * aws;
#pragma unroll
        for (int nc = 0; nc < 8; ++nc) {
#pragma unroll
            for (int kc = 0; kc < KCN; ++kc) {
                v16bf bm = bmp[(size_t)(((h * KCN + kc) * 8 + nc) * 32) + lane];
                acc[nc] = __builtin_amdgcn_wmma_f32_16x16x32_bf16(false, sa[kc], false, bm,
                                                                  (short)0, acc[nc], false, false);
            }
        }
    }
#pragma unroll
    for (int nc = 0; nc < 8; ++nc) {
        int col = nc * 16 + lcol;
#pragma unroll
        for (int r = 0; r < 8; ++r) {
            int M = r + lhi * 8;
            unsafeAtomicAdd(&agg[(size_t)idxS[wv][M] * DIMC + col], acc[nc][r]);
        }
    }
}

// ---------------- layernorm: z = LN(z + agg)
__global__ void k_norm(float* z, const float* agg, const float* g, const float* b) {
    int row = blockIdx.x;
    int t = threadIdx.x;
    __shared__ float red[128];
    float v = z[(size_t)row * 128 + t] + agg[(size_t)row * 128 + t];
    red[t] = v;
    __syncthreads();
    for (int s2 = 64; s2 > 0; s2 >>= 1) { if (t < s2) red[t] += red[t + s2]; __syncthreads(); }
    float mu = red[0] * (1.f / 128.f);
    __syncthreads();
    float d = v - mu;
    red[t] = d * d;
    __syncthreads();
    for (int s2 = 64; s2 > 0; s2 >>= 1) { if (t < s2) red[t] += red[t + s2]; __syncthreads(); }
    float var = red[0] * (1.f / 128.f);
    z[(size_t)row * 128 + t] = d * rsqrtf(var + 1e-5f) * g[t] + b[t];
}

static inline int cdiv(long long a, int b) { return (int)((a + b - 1) / b); }

extern "C" void kernel_launch(void* const* d_in, const int* in_sizes, int n_in,
                              void* d_out, int out_size, void* d_ws, size_t ws_size,
                              hipStream_t stream) {
    const int*   wid  = (const int*)d_in[0];
    const int*   tidv = (const int*)d_in[1];
    const int*   oid  = (const int*)d_in[2];
    const float* option_x = (const float*)d_in[3];
    const float* xw = (const float*)d_in[4];
    const float* xt = (const float*)d_in[5];
    const float* Wq[2] = { (const float*)d_in[6],  (const float*)d_in[12] };
    const float* bq[2] = { (const float*)d_in[7],  (const float*)d_in[13] };
    const float* Wk[2] = { (const float*)d_in[8],  (const float*)d_in[14] };
    const float* bk[2] = { (const float*)d_in[9],  (const float*)d_in[15] };
    const float* Wv[2] = { (const float*)d_in[10], (const float*)d_in[16] };
    const float* bv[2] = { (const float*)d_in[11], (const float*)d_in[17] };
    const float* Wo[2] = { (const float*)d_in[18], (const float*)d_in[20] };
    const float* bo[2] = { (const float*)d_in[19], (const float*)d_in[21] };
    const float* ln_g = (const float*)d_in[22];
    const float* ln_b = (const float*)d_in[23];

    float* out = (float*)d_out;
    float* zw = out;
    float* zt = out + (size_t)NWC * DIMC;
    float* zo = out + (size_t)(NWC + NTC) * DIMC;

    // workspace carve-up
    char* base = (char*)d_ws;
    size_t off = 0;
    auto carve = [&](size_t bytes) -> char* {
        char* p = base + off;
        off = (off + bytes + 255) & ~(size_t)255;
        return p;
    };
    __bf16* zedgeA = (__bf16*)carve((size_t)TILES * KCN * 512 * 2);   // 38.4 MB
    __bf16* Bq_b   = (__bf16*)carve((size_t)KCN * NCGN * 512 * 2);    // 768 KB
    __bf16* Bk_b   = (__bf16*)carve((size_t)KCN * NCGN * 512 * 2);
    __bf16* BM_b   = (__bf16*)carve((size_t)HEADC * KCN * 8 * 512 * 2);
    float*  c2buf  = (float*)carve((size_t)HEADC * DIMC * 4);
    float*  score  = (float*)carve((size_t)EC * HEADC * 4);
    unsigned* mbuf = (unsigned*)carve((size_t)NWC * HEADC * 4);
    float*  sbuf   = (float*)carve((size_t)NWC * HEADC * 4);
    float*  agg    = (float*)carve((size_t)NWC * DIMC * 4);

    k_init<<<cdiv((long long)(NWC + NTC + NOC) * DIMC, 256), 256, 0, stream>>>(out, xw, xt, option_x);

    const int ehGrid = cdiv((long long)EC * HEADC, 256);
    const int tileGrid = cdiv(TILES, 8);

    for (int ly = 0; ly < 2; ++ly) {
        k_pass0<<<cdiv((long long)TILES * KCN * 32, 256), 256, 0, stream>>>(
            zedgeA, zw, zt, zo, wid, tidv, oid);

        for (int br = 0; br < 2; ++br) {
            const float* pWq = Wq[br] + (size_t)ly * 384 * 1024;
            const float* pWk = Wk[br] + (size_t)ly * 384 * 1024;
            const float* pWv = Wv[br] + (size_t)ly * 384 * 1024;
            const float* pbq = bq[br] + (size_t)ly * 1024;
            const float* pbk = bk[br] + (size_t)ly * 1024;
            const float* pbv = bv[br] + (size_t)ly * 1024;
            const float* pWo = Wo[br] + (size_t)ly * 1024 * 128;
            const float* pbo = bo[br] + (size_t)ly * 128;
            const int*   idx = br ? tidv : wid;
            const int    nseg = br ? NTC : NWC;
            float*       zb = br ? zt : zw;

            k_convW<<<cdiv(2LL * KCN * NCGN * 32, 256), 256, 0, stream>>>(Bq_b, Bk_b, pWq, pWk);
            k_convM<<<cdiv((long long)HEADC * KCN * 8 * 32 * 16, 256), 256, 0, stream>>>(BM_b, pWv, pWo);
            k_c2<<<cdiv(HEADC * DIMC, 256), 256, 0, stream>>>(c2buf, pbv, pWo);

            k_fill<<<cdiv((long long)nseg * HEADC, 256), 256, 0, stream>>>(mbuf, (size_t)nseg * HEADC, 0x007FFFFFu); // map(-inf)
            k_fill<<<cdiv((long long)nseg * HEADC, 256), 256, 0, stream>>>((unsigned*)sbuf, (size_t)nseg * HEADC, 0u);
            k_fill<<<cdiv((long long)nseg * DIMC, 256), 256, 0, stream>>>((unsigned*)agg, (size_t)nseg * DIMC, 0u);

            k_pass1<<<tileGrid, 256, 0, stream>>>(score, zedgeA, Bq_b, Bk_b, pbq, pbk);
            k_segmax<<<ehGrid, 256, 0, stream>>>(mbuf, score, idx);
            k_expsum<<<ehGrid, 256, 0, stream>>>(score, sbuf, mbuf, idx);
            k_attn<<<ehGrid, 256, 0, stream>>>(score, sbuf, idx);
            k_pass3<<<tileGrid, 256, 0, stream>>>(agg, score, zedgeA, BM_b, c2buf, pbo, idx);
            k_norm<<<nseg, 128, 0, stream>>>(zb, agg, ln_g, ln_b);
        }
    }
}
